// RNN_30992484008226
// MI455X (gfx1250) — matrix operand order repaired and checked by reading, not compiled, not run on hardware
//
#include <hip/hip_runtime.h>
#include <hip/hip_bf16.h>
#include <math.h>

typedef _Float16 half_t;
typedef __attribute__((ext_vector_type(2)))  _Float16 v2h;
typedef __attribute__((ext_vector_type(8)))  _Float16 v8h;
typedef __attribute__((ext_vector_type(16))) _Float16 v16h;
typedef __attribute__((ext_vector_type(8)))  float    v8f;

#define BATCH   524288
#define TSTEPS  6
#define IFEAT   4
#define HID     20
#define GATES   80            // 4*HID
#define WAVES   4
#define RPW     16            // rows (batch elems) per wave = WMMA M
#define RPB     (WAVES*RPW)   // rows per block

// ws layout (floats): [0..5] sum_x per t, [6..11] sumsq_x per t,
// [12] sum_h2t5, [13] sumsq_h2t5, [16..] h2@t5 buffer (BATCH*HID floats)
#define WS_H2_OFF 16

// ---- branch-free activations (gfx1250 hardware TANH trans op) ----
__device__ __forceinline__ float fast_tanh(float x) {
#if __has_builtin(__builtin_amdgcn_tanhf)
    return __builtin_amdgcn_tanhf(x);
#else
    float e = __expf(2.0f * x);
    return 1.0f - 2.0f * __builtin_amdgcn_rcpf(e + 1.0f);
#endif
}

__device__ __forceinline__ float fast_sigmoid(float x) {
#if __has_builtin(__builtin_amdgcn_tanhf)
    return 0.5f * __builtin_amdgcn_tanhf(0.5f * x) + 0.5f;
#else
    return __builtin_amdgcn_rcpf(1.0f + __expf(-x));
#endif
}

// pack 2 f32 -> 2 f16 (v_cvt_pk_rtz_f16_f32); bit-cast __fp16x2 -> _Float16x2
__device__ __forceinline__ v2h pack2h(float a, float b) {
    return __builtin_bit_cast(v2h, __builtin_amdgcn_cvt_pkrtz(a, b));
}

// A fragment: 16x32 f16, A-layout (lanes 0-15: rows, K 0..7 & 16..23;
// lanes 16-31: same rows, K 8..15 & 24..31). base points at row-major [16][rowStrideH].
__device__ __forceinline__ v16h load_a_frag(const half_t* base, int rowStrideH, int lane) {
    int m  = lane & 15;
    int hi = lane >> 4;
    const half_t* p = base + m * rowStrideH + hi * 8;
    v8h lo  = *(const v8h*)(p);
    v8h hiv = *(const v8h*)(p + 16);
    return __builtin_shufflevector(lo, hiv, 0,1,2,3,4,5,6,7,8,9,10,11,12,13,14,15);
}

// B fragment: 32x16 f16, B-layout (lane = column N; lanes 0-15 hold K 0..7 & 16..23,
// lanes 16-31 hold K 8..15 & 24..31). Weights stored row-major [N=80][rowStrideH] (K contiguous).
__device__ __forceinline__ v16h load_b_frag(const half_t* wbase, int rowStrideH,
                                            int lane, int tile, int kchunk) {
    int n     = lane & 15;
    int khalf = (lane >> 4) * 8;
    const half_t* p = wbase + (tile * 16 + n) * rowStrideH + kchunk + khalf;
    v8h lo  = *(const v8h*)(p);
    v8h hiv = *(const v8h*)(p + 16);
    return __builtin_shufflevector(lo, hiv, 0,1,2,3,4,5,6,7,8,9,10,11,12,13,14,15);
}

__global__ void zero_ws_kernel(float* ws) {
    if (threadIdx.x < WS_H2_OFF) ws[threadIdx.x] = 0.0f;
}

// Kernel A: per-channel (t) sum / sumsq of x over (B, I).
__global__ __launch_bounds__(256) void xstats_kernel(const float* __restrict__ x,
                                                     float* __restrict__ ws) {
    int b = blockIdx.x * 256 + threadIdx.x;
    const float4* xv = (const float4*)(x + (size_t)b * 24);
    float s[TSTEPS], s2[TSTEPS];
#pragma unroll
    for (int t = 0; t < TSTEPS; ++t) {
        float4 q = xv[t];
        s[t]  = q.x + q.y + q.z + q.w;
        s2[t] = q.x * q.x + q.y * q.y + q.z * q.z + q.w * q.w;
    }
#pragma unroll
    for (int t = 0; t < TSTEPS; ++t) {
#pragma unroll
        for (int off = 16; off > 0; off >>= 1) {
            s[t]  += __shfl_down(s[t],  off, 32);
            s2[t] += __shfl_down(s2[t], off, 32);
        }
    }
    __shared__ float bs[12];
    if (threadIdx.x < 12) bs[threadIdx.x] = 0.0f;
    __syncthreads();
    if ((threadIdx.x & 31) == 0) {
#pragma unroll
        for (int t = 0; t < TSTEPS; ++t) {
            atomicAdd(&bs[t], s[t]);
            atomicAdd(&bs[6 + t], s2[t]);
        }
    }
    __syncthreads();
    if (threadIdx.x < 12) atomicAdd(&ws[threadIdx.x], bs[threadIdx.x]);
}

// Kernel B: BN(x) + 2-layer LSTM with WMMA gate matmuls; emits h2@t5 + its stats.
__global__ __launch_bounds__(WAVES * 32) void lstm_kernel(
    const float* __restrict__ x,
    const float* __restrict__ gamma, const float* __restrict__ beta,
    const float* __restrict__ W_ih0, const float* __restrict__ W_hh0,
    const float* __restrict__ b_ih0, const float* __restrict__ b_hh0,
    const float* __restrict__ W_ih1, const float* __restrict__ W_hh1,
    const float* __restrict__ b_ih1, const float* __restrict__ b_hh1,
    float* __restrict__ ws) {
    // Block-shared weights (gate-interleaved column order n = 4*hu + gate)
    __shared__ __align__(16) half_t w0h[GATES * 32];   // [n][k] k: 0..3 x, 4..23 h1, pad
    __shared__ __align__(16) half_t w1h[GATES * 64];   // [n][k] k: 0..19 h1, 32..51 h2, pad
    __shared__ float bias0s[GATES];
    __shared__ float bias1s[GATES];
    // Per-wave staging
    __shared__ __align__(16) half_t st0s[WAVES][RPW * 32];
    __shared__ __align__(16) half_t st1s[WAVES][RPW * 64];
    __shared__ __align__(16) half_t xshs[WAVES][RPW * 24];
    __shared__ __align__(16) float  gbufs[WAVES][RPW * GATES];

    const int tid  = threadIdx.x;
    const int lane = tid & 31;
    const int wave = tid >> 5;
    const int m    = lane & 15;
    const int hi   = lane >> 4;
    const int hu0  = 10 * hi;              // this lane's first hidden unit
    const int rowBase = blockIdx.x * RPB + wave * RPW;

    // ---- load + repack weights into LDS ----
    for (int idx = tid; idx < GATES * 32; idx += blockDim.x) {
        int n = idx >> 5, k = idx & 31;
        int gate = n & 3, hu = n >> 2;
        int r = gate * HID + hu;                 // original torch row (i,f,g,o blocks)
        float w = 0.0f;
        if (k < IFEAT)            w = W_ih0[r * IFEAT + k];
        else if (k < IFEAT + HID) w = W_hh0[r * HID + (k - IFEAT)];
        w0h[idx] = (half_t)w;
    }
    for (int idx = tid; idx < GATES * 64; idx += blockDim.x) {
        int n = idx >> 6, k = idx & 63;
        int gate = n & 3, hu = n >> 2;
        int r = gate * HID + hu;
        float w = 0.0f;
        if (k < HID)                       w = W_ih1[r * HID + k];
        else if (k >= 32 && k < 32 + HID)  w = W_hh1[r * HID + (k - 32)];
        w1h[idx] = (half_t)w;
    }
    for (int n = tid; n < GATES; n += blockDim.x) {
        int gate = n & 3, hu = n >> 2;
        int r = gate * HID + hu;
        bias0s[n] = b_ih0[r] + b_hh0[r];
        bias1s[n] = b_ih1[r] + b_hh1[r];
    }

    // ---- BN(x) parameters from kernel-A stats ----
    float meanx[TSTEPS], scalex[TSTEPS], shiftx[TSTEPS];
    const float invNx = 1.0f / ((float)BATCH * (float)IFEAT);
    for (int t = 0; t < TSTEPS; ++t) {
        float mu   = ws[t] * invNx;
        float var  = ws[6 + t] * invNx - mu * mu;
        float rstd = rsqrtf(var + 1e-5f);
        meanx[t]  = mu;
        scalex[t] = rstd * gamma[t];
        shiftx[t] = beta[t];
    }

    half_t* st0 = st0s[wave];
    half_t* st1 = st1s[wave];
    half_t* xw  = xshs[wave];
    float*  gb  = gbufs[wave];

    // normalized x tile preload (f16)
    for (int q = lane; q < RPW * 24; q += 32) {
        int mm = q / 24, j = q % 24;
        int t = j >> 2;
        float v = x[(size_t)(rowBase + mm) * 24 + j];
        xw[q] = (half_t)((v - meanx[t]) * scalex[t] + shiftx[t]);
    }
    // zero-pad staging K regions (written once, never touched again)
    for (int q = lane; q < RPW * 8;  q += 32) { int mm = q >> 3, k = 24 + (q & 7);  st0[mm * 32 + k] = (half_t)0.0f; }
    for (int q = lane; q < RPW * 12; q += 32) { int mm = q / 12, k = 20 + (q % 12); st1[mm * 64 + k] = (half_t)0.0f; }
    for (int q = lane; q < RPW * 12; q += 32) { int mm = q / 12, k = 52 + (q % 12); st1[mm * 64 + k] = (half_t)0.0f; }
    __syncthreads();

    float bias0l[5], bias1l[5];
#pragma unroll
    for (int nt = 0; nt < 5; ++nt) {
        bias0l[nt] = bias0s[16 * nt + m];
        bias1l[nt] = bias1s[16 * nt + m];
    }

    // per-lane state: 10 cells (row m, hu = hu0 + j, j = 0..9)
    float h1[10], c1[10], h2[10], c2[10];
#pragma unroll
    for (int j = 0; j < 10; ++j) { h1[j] = c1[j] = h2[j] = c2[j] = 0.0f; }

    for (int t = 0; t < TSTEPS; ++t) {
        // ---- stage layer-0 A: [x_t | h1_prev | 0] ----
        for (int q = lane; q < 64; q += 32) {
            int mm = q >> 2, i = q & 3;
            st0[mm * 32 + i] = xw[mm * 24 + t * 4 + i];
        }
#pragma unroll
        for (int k = 0; k < 5; ++k)   // 10 contiguous halfs starting at col 4+hu0
            *(v2h*)&st0[m * 32 + 4 + hu0 + 2 * k] = pack2h(h1[2 * k], h1[2 * k + 1]);
        __syncthreads();
        // ---- layer-0 WMMA: gates = bias + [x|h1] @ Wcat0 ----
        {
            v16h a = load_a_frag(st0, 32, lane);
#pragma unroll
            for (int nt = 0; nt < 5; ++nt) {
                v8f acc;
#pragma unroll
                for (int e = 0; e < 8; ++e) acc[e] = bias0l[nt];
                v16h b = load_b_frag(w0h, 32, lane, nt, 0);
                acc = __builtin_amdgcn_wmma_f32_16x16x32_f16(false, a, false, b,
                                                             (short)0, acc, false, false);
#pragma unroll
                for (int r = 0; r < 8; ++r)
                    gb[(r + 8 * hi) * GATES + 16 * nt + m] = acc[r];
            }
        }
        __syncthreads();
        // ---- layer-0 nonlinearity / state update ----
#pragma unroll
        for (int j = 0; j < 10; ++j) {
            float4 g4 = *(const float4*)&gb[m * GATES + 4 * (hu0 + j)];
            float ig = fast_sigmoid(g4.x), fg = fast_sigmoid(g4.y);
            float gg = fast_tanh(g4.z),    og = fast_sigmoid(g4.w);
            c1[j] = fg * c1[j] + ig * gg;
            h1[j] = og * fast_tanh(c1[j]);
        }
        // ---- stage layer-1 A: [h1_t | 0] and [h2_prev | 0] ----
#pragma unroll
        for (int k = 0; k < 5; ++k) {
            *(v2h*)&st1[m * 64 + hu0 + 2 * k]      = pack2h(h1[2 * k], h1[2 * k + 1]);
            *(v2h*)&st1[m * 64 + 32 + hu0 + 2 * k] = pack2h(h2[2 * k], h2[2 * k + 1]);
        }
        __syncthreads();
        // ---- layer-1 WMMA: gates = bias + h1@W_ih1^T + h2@W_hh1^T ----
        {
            v16h a0 = load_a_frag(st1,      64, lane);
            v16h a1 = load_a_frag(st1 + 32, 64, lane);
#pragma unroll
            for (int nt = 0; nt < 5; ++nt) {
                v8f acc;
#pragma unroll
                for (int e = 0; e < 8; ++e) acc[e] = bias1l[nt];
                v16h b0 = load_b_frag(w1h, 64, lane, nt, 0);
                acc = __builtin_amdgcn_wmma_f32_16x16x32_f16(false, a0, false, b0,
                                                             (short)0, acc, false, false);
                v16h b1 = load_b_frag(w1h, 64, lane, nt, 32);
                acc = __builtin_amdgcn_wmma_f32_16x16x32_f16(false, a1, false, b1,
                                                             (short)0, acc, false, false);
#pragma unroll
                for (int r = 0; r < 8; ++r)
                    gb[(r + 8 * hi) * GATES + 16 * nt + m] = acc[r];
            }
        }
        __syncthreads();
        // ---- layer-1 nonlinearity / state update ----
#pragma unroll
        for (int j = 0; j < 10; ++j) {
            float4 g4 = *(const float4*)&gb[m * GATES + 4 * (hu0 + j)];
            float ig = fast_sigmoid(g4.x), fg = fast_sigmoid(g4.y);
            float gg = fast_tanh(g4.z),    og = fast_sigmoid(g4.w);
            c2[j] = fg * c2[j] + ig * gg;
            h2[j] = og * fast_tanh(c2[j]);
        }
        // next iteration's first __syncthreads orders these reads vs. new gb writes
    }

    // ---- epilogue: store h2@t5 (10 contiguous floats/lane), accumulate t=5 stats ----
    float s = 0.0f, s2 = 0.0f;
    {
        float* hdst = ws + WS_H2_OFF + (size_t)(rowBase + m) * HID + hu0;
#pragma unroll
        for (int j = 0; j < 10; ++j) {
            float v = h2[j];
            hdst[j] = v;
            s += v;
            s2 += v * v;
        }
    }
#pragma unroll
    for (int off = 16; off > 0; off >>= 1) {
        s  += __shfl_down(s,  off, 32);
        s2 += __shfl_down(s2, off, 32);
    }
    if (lane == 0) {
        atomicAdd(&ws[12], s);
        atomicAdd(&ws[13], s2);
    }
}

// Kernel C: out[b] = sum_hu BN5(h2[b,hu]) * W_out[hu] + b_out
__global__ __launch_bounds__(256) void finalize_kernel(const float* __restrict__ ws,
                                                       const float* __restrict__ gamma,
                                                       const float* __restrict__ beta,
                                                       const float* __restrict__ W_out,
                                                       const float* __restrict__ b_out,
                                                       float* __restrict__ out) {
    int b = blockIdx.x * 256 + threadIdx.x;
    const float invN = 1.0f / ((float)BATCH * (float)HID);
    float mu   = ws[12] * invN;
    float var  = ws[13] * invN - mu * mu;
    float rstd = rsqrtf(var + 1e-5f);
    float g5 = gamma[5] * rstd, b5 = beta[5];
    const float* h = ws + WS_H2_OFF + (size_t)b * HID;
    float acc = b_out[0];
#pragma unroll
    for (int k = 0; k < HID; ++k)
        acc += ((h[k] - mu) * g5 + b5) * W_out[k];
    out[b] = acc;
}

extern "C" void kernel_launch(void* const* d_in, const int* in_sizes, int n_in,
                              void* d_out, int out_size, void* d_ws, size_t ws_size,
                              hipStream_t stream) {
    const float* x     = (const float*)d_in[0];
    const float* gamma = (const float*)d_in[1];
    const float* beta  = (const float*)d_in[2];
    const float* W_ih0 = (const float*)d_in[3];
    const float* W_hh0 = (const float*)d_in[4];
    const float* b_ih0 = (const float*)d_in[5];
    const float* b_hh0 = (const float*)d_in[6];
    const float* W_ih1 = (const float*)d_in[7];
    const float* W_hh1 = (const float*)d_in[8];
    const float* b_ih1 = (const float*)d_in[9];
    const float* b_hh1 = (const float*)d_in[10];
    const float* W_out = (const float*)d_in[11];
    const float* b_out = (const float*)d_in[12];
    float* out = (float*)d_out;
    float* ws  = (float*)d_ws;

    zero_ws_kernel<<<1, 32, 0, stream>>>(ws);
    xstats_kernel<<<BATCH / 256, 256, 0, stream>>>(x, ws);
    lstm_kernel<<<BATCH / RPB, WAVES * 32, 0, stream>>>(
        x, gamma, beta, W_ih0, W_hh0, b_ih0, b_hh0,
        W_ih1, W_hh1, b_ih1, b_hh1, ws);
    finalize_kernel<<<BATCH / 256, 256, 0, stream>>>(ws, gamma, beta, W_out, b_out, out);
}
